// RModel_88648124990070
// MI455X (gfx1250) — matrix operand order repaired and checked by reading, not compiled, not run on hardware
//
#include <hip/hip_runtime.h>

#define N_NODES   100000
#define N_EDGES   1200000
#define N_GRAPHS  512
#define N_REL     3
#define NUM_CLASS 10

typedef __attribute__((ext_vector_type(16))) _Float16 v16h;
typedef __attribute__((ext_vector_type(8)))  float    v8f;

// ---------------------------------------------------------------------------
// A-tile loader: 16(M) x 32(K) f16 per ISA 7.12.2.
// Lane L holds row M = L%16; half t maps to K = (t/8)*16 + (L/16)*8 + (t%8),
// i.e. two contiguous 8-float runs -> 4x float4 loads + pack-converts.
// ---------------------------------------------------------------------------
__device__ __forceinline__ v16h load_a_tile(const float* __restrict__ h, int ld,
                                            int node0, int k0, int lane) {
  const int row = lane & 15;
  const int hi  = lane >> 4;
  const float* p = h + (size_t)(node0 + row) * ld + k0 + (hi << 3);
  const float4 f0 = *(const float4*)(p + 0);
  const float4 f1 = *(const float4*)(p + 4);
  const float4 f2 = *(const float4*)(p + 16);
  const float4 f3 = *(const float4*)(p + 20);
  v16h a;
  a[0]  = (_Float16)f0.x; a[1]  = (_Float16)f0.y;
  a[2]  = (_Float16)f0.z; a[3]  = (_Float16)f0.w;
  a[4]  = (_Float16)f1.x; a[5]  = (_Float16)f1.y;
  a[6]  = (_Float16)f1.z; a[7]  = (_Float16)f1.w;
  a[8]  = (_Float16)f2.x; a[9]  = (_Float16)f2.y;
  a[10] = (_Float16)f2.z; a[11] = (_Float16)f2.w;
  a[12] = (_Float16)f3.x; a[13] = (_Float16)f3.y;
  a[14] = (_Float16)f3.z; a[15] = (_Float16)f3.w;
  return a;
}

// ---------------------------------------------------------------------------
// Weight pre-pack: convert W[K, ncol] f32 into per-lane WMMA B layout
// (32(K) x 16(N) f16: lane L holds col N = L%16, half t -> K = (t/8)*16 + (L/16)*8 + (t%8)),
// zero-padding K >= kReal.  One thread per (chunk, cbIdx, lane) writes 16 halves.
// Tile order: ((chunk * nCb + cbIdx) * 32 + lane) * 16.
// ---------------------------------------------------------------------------
__global__ void pack_b_kernel(const float* __restrict__ W, _Float16* __restrict__ out,
                              int kReal, int ncol, int kchunks) {
  const int t    = blockIdx.x * blockDim.x + threadIdx.x;
  const int nCb  = ncol >> 4;
  const int tot  = kchunks * nCb * 32;
  if (t >= tot) return;
  const int lane  = t & 31;
  const int cbIdx = (t >> 5) % nCb;
  const int chunk = (t >> 5) / nCb;
  const int col   = (cbIdx << 4) + (lane & 15);
  const int kbase = (chunk << 5) + ((lane >> 4) << 3);
  _Float16* o = out + (size_t)t * 16;
#pragma unroll
  for (int j = 0; j < 16; ++j) {
    const int k = kbase + j + ((j >> 3) << 3);
    const int kc = k < kReal ? k : 0;               // keep load in-bounds
    const float v = W[(size_t)kc * ncol + col];
    o[j] = (_Float16)((k < kReal) ? v : 0.0f);
  }
}

// ---------------------------------------------------------------------------
// Per-(rel,dst) in-degree counts (shared by both R-GCN layers)
// ---------------------------------------------------------------------------
__global__ void count_kernel(const int* __restrict__ ei, const int* __restrict__ et,
                             float* __restrict__ cnt) {
  const int e = blockIdx.x * blockDim.x + threadIdx.x;
  if (e >= N_EDGES) return;
  const int dst = ei[N_EDGES + e];
  const int r   = et[e];
  atomicAdd(&cnt[(size_t)r * N_NODES + dst], 1.0f);
}

// ---------------------------------------------------------------------------
// Embedding gather + pre-MLP:  h0 = relu(concat(se[x0], ce[x1]) @ pre_w + b)
// One wave = 16 nodes.  K=16 (zero-padded to 32 in A and packed B), N=32.
// ---------------------------------------------------------------------------
__global__ void embed_pre_kernel(const int* __restrict__ x,
                                 const float* __restrict__ se,
                                 const float* __restrict__ ce,
                                 const _Float16* __restrict__ Wp,   // packed pre_w
                                 const float* __restrict__ pre_b,
                                 float* __restrict__ h0, int nNodes) {
  const int wave  = (blockIdx.x * blockDim.x + threadIdx.x) >> 5;
  const int lane  = threadIdx.x & 31;
  const int node0 = wave << 4;
  if (node0 >= nNodes) return;          // wave-uniform: EXEC stays all-1s
  const int row = lane & 15, hi = lane >> 4;
  const int node = node0 + row;
  const int sidx = x[2 * node], cidx = x[2 * node + 1];
  // lanes 0-15 supply K=0..7 (shape emb), lanes 16-31 supply K=8..15 (color emb)
  const float* emb = hi ? (ce + cidx * 8) : (se + sidx * 8);
  const float4 e0 = *(const float4*)(emb + 0);
  const float4 e1 = *(const float4*)(emb + 4);
  v16h a;
  a[0] = (_Float16)e0.x; a[1] = (_Float16)e0.y;
  a[2] = (_Float16)e0.z; a[3] = (_Float16)e0.w;
  a[4] = (_Float16)e1.x; a[5] = (_Float16)e1.y;
  a[6] = (_Float16)e1.z; a[7] = (_Float16)e1.w;
#pragma unroll
  for (int t = 8; t < 16; ++t) a[t] = (_Float16)0.0f;   // K=16..31 zero pad
#pragma unroll
  for (int cbIdx = 0; cbIdx < 2; ++cbIdx) {
    const v16h b = *(const v16h*)(Wp + ((size_t)(cbIdx * 32) + lane) * 16);
    v8f acc = {};
    acc = __builtin_amdgcn_wmma_f32_16x16x32_f16(false, a, false, b,
                                                 (short)0, acc, false, false);
    const int col  = (cbIdx << 4) + row;
    const float bv = pre_b[col];
#pragma unroll
    for (int v = 0; v < 8; ++v) {
      const int m = v + (hi << 3);      // C layout: VGPR v, M = v + 8*(lane/16)
      h0[(size_t)(node0 + m) * 32 + col] = fmaxf(acc[v] + bv, 0.0f);
    }
  }
}

// ---------------------------------------------------------------------------
// Node-level GEMM:  out[N,64] = h[N,CIN] @ W[CIN,64] (+ bias).
// One wave = 16 nodes; B comes pre-packed (v16h per lane, 2x b128 loads).
// ---------------------------------------------------------------------------
template<int CIN, bool BIAS>
__global__ void gemm_kernel(const float* __restrict__ h, const _Float16* __restrict__ Wp,
                            const float* __restrict__ bias, float* __restrict__ out,
                            int nNodes) {
  const int wave  = (blockIdx.x * blockDim.x + threadIdx.x) >> 5;
  const int lane  = threadIdx.x & 31;
  const int node0 = wave << 4;
  if (node0 >= nNodes) return;          // wave-uniform
  const int lo = lane & 15, hi = lane >> 4;
  constexpr int KC = CIN / 32;
  v16h a[KC];
#pragma unroll
  for (int c = 0; c < KC; ++c) a[c] = load_a_tile(h, CIN, node0, c * 32, lane);
#pragma unroll
  for (int cbIdx = 0; cbIdx < 4; ++cbIdx) {
    v8f acc = {};
#pragma unroll
    for (int c = 0; c < KC; ++c) {
      const v16h b = *(const v16h*)(Wp + ((size_t)((c * 4 + cbIdx) * 32) + lane) * 16);
      acc = __builtin_amdgcn_wmma_f32_16x16x32_f16(false, a[c], false, b,
                                                   (short)0, acc, false, false);
    }
    const float bv = BIAS ? bias[(cbIdx << 4) + lo] : 0.0f;
#pragma unroll
    for (int v = 0; v < 8; ++v)
      out[(size_t)(node0 + v + (hi << 3)) * 64 + (cbIdx << 4) + lo] = acc[v] + bv;
  }
}

// ---------------------------------------------------------------------------
// Edge scatter (mean folded in via precomputed counts):
//   hnext[dst] += hr[src] / max(cnt[rel][dst],1)  for edges of `rel`
// 16 threads per edge, float4 gather + 4 float atomics each.
// ---------------------------------------------------------------------------
__global__ void scatter_kernel(const int* __restrict__ ei, const int* __restrict__ et,
                               const float* __restrict__ hr, const float* __restrict__ cnt,
                               float* __restrict__ hnext, int rel) {
  const int t = blockIdx.x * blockDim.x + threadIdx.x;
  const int e = t >> 4;
  if (e >= N_EDGES) return;
  if (et[e] != rel) return;
  const int q   = t & 15;
  const int src = ei[e];
  const int dst = ei[N_EDGES + e];
  const float s = 1.0f / fmaxf(cnt[(size_t)rel * N_NODES + dst], 1.0f);
  const float4 v = *(const float4*)(hr + (size_t)src * 64 + (q << 2));
  float* o = hnext + (size_t)dst * 64 + (q << 2);
  atomicAdd(o + 0, v.x * s);
  atomicAdd(o + 1, v.y * s);
  atomicAdd(o + 2, v.z * s);
  atomicAdd(o + 3, v.w * s);
}

__global__ void relu_kernel(float* __restrict__ p, int n) {
  const int i = blockIdx.x * blockDim.x + threadIdx.x;
  if (i < n) p[i] = fmaxf(p[i], 0.0f);
}

// ---------------------------------------------------------------------------
// Graph mean-pool (atomic sums + counts) and classifier
// ---------------------------------------------------------------------------
__global__ void pool_kernel(const float* __restrict__ h, const int* __restrict__ batch,
                            float* __restrict__ pool, float* __restrict__ gcnt) {
  const int t = blockIdx.x * blockDim.x + threadIdx.x;
  const int node = t >> 4;
  if (node >= N_NODES) return;
  const int q = t & 15;
  const int g = batch[node];
  const float4 v = *(const float4*)(h + (size_t)node * 64 + (q << 2));
  float* o = pool + (size_t)g * 64 + (q << 2);
  atomicAdd(o + 0, v.x);
  atomicAdd(o + 1, v.y);
  atomicAdd(o + 2, v.z);
  atomicAdd(o + 3, v.w);
  if (q == 0) atomicAdd(&gcnt[g], 1.0f);
}

__global__ void classify_kernel(const float* __restrict__ pool, const float* __restrict__ gcnt,
                                const float* __restrict__ cls_w, const float* __restrict__ cls_b,
                                float* __restrict__ out) {
  const int t = blockIdx.x * blockDim.x + threadIdx.x;
  if (t >= N_GRAPHS * NUM_CLASS) return;
  const int g = t / NUM_CLASS, c = t % NUM_CLASS;
  const float inv = 1.0f / fmaxf(gcnt[g], 1.0f);
  float acc = cls_b[c];
#pragma unroll 8
  for (int f = 0; f < 64; ++f)
    acc += pool[(size_t)g * 64 + f] * inv * cls_w[f * NUM_CLASS + c];
  out[t] = acc;
}

// ---------------------------------------------------------------------------
extern "C" void kernel_launch(void* const* d_in, const int* in_sizes, int n_in,
                              void* d_out, int out_size, void* d_ws, size_t ws_size,
                              hipStream_t stream) {
  const int*   x     = (const int*)  d_in[0];
  const int*   ei    = (const int*)  d_in[1];
  const int*   et    = (const int*)  d_in[2];
  const int*   batch = (const int*)  d_in[3];
  const float* se    = (const float*)d_in[4];
  const float* ce    = (const float*)d_in[5];
  const float* pre_w = (const float*)d_in[6];
  const float* pre_b = (const float*)d_in[7];
  const float* w1    = (const float*)d_in[8];
  const float* root1 = (const float*)d_in[9];
  const float* b1    = (const float*)d_in[10];
  const float* w2    = (const float*)d_in[11];
  const float* root2 = (const float*)d_in[12];
  const float* b2    = (const float*)d_in[13];
  const float* cls_w = (const float*)d_in[14];
  const float* cls_b = (const float*)d_in[15];
  float* out = (float*)d_out;

  // Workspace layout (floats); all regions 16B+ aligned.
  float* ws   = (float*)d_ws;
  float* cnt  = ws;                                   // 3*N
  float* h0   = cnt  + (size_t)N_REL  * N_NODES;      // N*32
  float* hr   = h0   + (size_t)N_NODES * 32;          // N*64
  float* h1   = hr   + (size_t)N_NODES * 64;          // N*64
  float* h2   = h1   + (size_t)N_NODES * 64;          // N*64
  float* pool = h2   + (size_t)N_NODES * 64;          // G*64
  float* gcnt = pool + (size_t)N_GRAPHS * 64;         // G
  // packed f16 weight tiles (32B-aligned: gcnt region ends on 64B boundary)
  _Float16* pbPre   = (_Float16*)(gcnt + N_GRAPHS);   // 1*2*32*16  = 1024 halves
  _Float16* pbRoot1 = pbPre   + 1024;                 // 1*4*32*16  = 2048
  _Float16* pbW1    = pbRoot1 + 2048;                 // 3 * 2048
  _Float16* pbRoot2 = pbW1    + 3 * 2048;             // 2*4*32*16  = 4096
  _Float16* pbW2    = pbRoot2 + 4096;                 // 3 * 4096

  hipMemsetAsync(cnt,  0, (size_t)N_REL * N_NODES * sizeof(float), stream);
  hipMemsetAsync(pool, 0, (size_t)N_GRAPHS * 65 * sizeof(float), stream);

  // ---- pack weights into WMMA B layout (tiny; runs every launch) ----
  pack_b_kernel<<<1,  64, 0, stream>>>(pre_w, pbPre,   16, 32, 1);
  pack_b_kernel<<<1, 128, 0, stream>>>(root1, pbRoot1, 32, 64, 1);
  for (int r = 0; r < N_REL; ++r)
    pack_b_kernel<<<1, 128, 0, stream>>>(w1 + (size_t)r * 32 * 64,
                                         pbW1 + (size_t)r * 2048, 32, 64, 1);
  pack_b_kernel<<<1, 256, 0, stream>>>(root2, pbRoot2, 64, 64, 2);
  for (int r = 0; r < N_REL; ++r)
    pack_b_kernel<<<1, 256, 0, stream>>>(w2 + (size_t)r * 64 * 64,
                                         pbW2 + (size_t)r * 4096, 64, 64, 2);

  const int gemmBlocks = ((N_NODES / 16) * 32 + 255) / 256;   // 16 nodes / wave
  const int scatBlocks = (N_EDGES * 16 + 255) / 256;

  count_kernel<<<(N_EDGES + 255) / 256, 256, 0, stream>>>(ei, et, cnt);
  embed_pre_kernel<<<gemmBlocks, 256, 0, stream>>>(x, se, ce, pbPre, pre_b, h0, N_NODES);

  // ---- R-GCN layer 1 (CIN=32) ----
  gemm_kernel<32, true><<<gemmBlocks, 256, 0, stream>>>(h0, pbRoot1, b1, h1, N_NODES);
  for (int r = 0; r < N_REL; ++r) {
    gemm_kernel<32, false><<<gemmBlocks, 256, 0, stream>>>(h0, pbW1 + (size_t)r * 2048,
                                                           nullptr, hr, N_NODES);
    scatter_kernel<<<scatBlocks, 256, 0, stream>>>(ei, et, hr, cnt, h1, r);
  }
  relu_kernel<<<(N_NODES * 64 + 255) / 256, 256, 0, stream>>>(h1, N_NODES * 64);

  // ---- R-GCN layer 2 (CIN=64) ----
  gemm_kernel<64, true><<<gemmBlocks, 256, 0, stream>>>(h1, pbRoot2, b2, h2, N_NODES);
  for (int r = 0; r < N_REL; ++r) {
    gemm_kernel<64, false><<<gemmBlocks, 256, 0, stream>>>(h1, pbW2 + (size_t)r * 4096,
                                                           nullptr, hr, N_NODES);
    scatter_kernel<<<scatBlocks, 256, 0, stream>>>(ei, et, hr, cnt, h2, r);
  }
  relu_kernel<<<(N_NODES * 64 + 255) / 256, 256, 0, stream>>>(h2, N_NODES * 64);

  // ---- pool + classify ----
  pool_kernel<<<(N_NODES * 16 + 255) / 256, 256, 0, stream>>>(h2, batch, pool, gcnt);
  classify_kernel<<<(N_GRAPHS * NUM_CLASS + 255) / 256, 256, 0, stream>>>(
      pool, gcnt, cls_w, cls_b, out);
}
